// QKVAttention_23433341567365
// MI455X (gfx1250) — matrix operand order repaired and checked by reading, not compiled, not run on hardware
//
#include <hip/hip_runtime.h>
#include <hip/hip_bf16.h>

#define HEADS 8
#define C     64
#define T     1024
#define S_ENC 1024
#define S_TOT (S_ENC + T)      // 2048
#define WAVES  4
#define BLOCK  (WAVES * 32)    // 128 threads = 4 wave32
#define TTILE  16              // query rows per wave
#define TBLK   (WAVES * TTILE) // 64 query rows per workgroup
#define SCHUNK 64              // kv columns per iteration
#define LROW   72              // LDS row stride in halves: 144 B, 16B-aligned

// scale = 1/sqrt(C) = 0.125, folded with log2(e); applied to Q at staging time
#define SCALE2 (0.125f * 1.44269504088896340736f)

typedef __attribute__((ext_vector_type(16))) __bf16 v16bf;
typedef __attribute__((ext_vector_type(8)))  __bf16 v8bf;
typedef __attribute__((ext_vector_type(4)))  __bf16 v4bf;
typedef __attribute__((ext_vector_type(8)))  float  v8f;
typedef __attribute__((ext_vector_type(4)))  float  v4f;

#if __has_builtin(__builtin_amdgcn_exp2f)
#define fast_exp2(x) __builtin_amdgcn_exp2f(x)
#else
#define fast_exp2(x) exp2f(x)
#endif

__device__ __forceinline__ v16bf combine8(v8bf lo, v8bf hi) {
    v16bf r;
#pragma unroll
    for (int i = 0; i < 8; ++i) { r[i] = lo[i]; r[i + 8] = hi[i]; }
    return r;
}
// A-operand fragment: two 16B chunks at half-offsets lo/hi within one row
__device__ __forceinline__ v16bf lds_frag(const __bf16* row, int lo, int hi) {
    return combine8(*(const v8bf*)(row + lo), *(const v8bf*)(row + hi));
}
// B-operand fragment: 16 contiguous halves (two aligned 16B loads)
__device__ __forceinline__ v16bf lds_frag16(const __bf16* p) {
    return combine8(*(const v8bf*)p, *(const v8bf*)(p + 8));
}

__device__ __forceinline__ v8f wmma_bf16(v16bf a, v16bf b, v8f c) {
    return __builtin_amdgcn_wmma_f32_16x16x32_bf16(false, a, false, b,
                                                   (short)0, c, false, false);
}

__global__ __launch_bounds__(BLOCK)
void qkv_attn_flash_bf16(const float* __restrict__ x,
                         const float* __restrict__ ekv,
                         float* __restrict__ out) {
    __shared__ __attribute__((aligned(16))) __bf16 Ks[SCHUNK][LROW];        // K chunk, [s][c]
    __shared__ __attribute__((aligned(16))) __bf16 Vt[C][LROW];             // V chunk, [c][s]
    __shared__ __attribute__((aligned(16))) __bf16 Qs[WAVES][TTILE][LROW];  // Q tile,  [t][c]
    __shared__ __attribute__((aligned(16))) __bf16 Ps[WAVES][TTILE][LROW];  // P tile,  [t][s]

    const int tid  = threadIdx.x;
    const int lane = tid & 31;
    const int w    = tid >> 5;
    const int bh   = blockIdx.y;
    const int bb   = bh / HEADS;
    const int hh   = bh % HEADS;
    const int t0w  = blockIdx.x * TBLK + w * TTILE;

    // x: (b, 3, H, C, T)   ekv: (b, 2, H, C, S_ENC)   out: (b, H, C, T)
    const float* qg  = x   + (size_t)(((bb * 3 + 0) * HEADS + hh) * C) * T;
    const float* kg  = x   + (size_t)(((bb * 3 + 1) * HEADS + hh) * C) * T;
    const float* vg  = x   + (size_t)(((bb * 3 + 2) * HEADS + hh) * C) * T;
    const float* ekg = ekv + (size_t)(((bb * 2 + 0) * HEADS + hh) * C) * S_ENC;
    const float* evg = ekv + (size_t)(((bb * 2 + 1) * HEADS + hh) * C) * S_ENC;
    float*       og  = out + (size_t)((bb * HEADS + hh) * C) * T;

    // ---- stage this wave's Q tile (16 t x 64 c), float4 loads, fp32->bf16 ----
    // softmax scale (exp2 domain) folded into Q here: scores emerge pre-scaled.
    for (int i = lane; i < C * (TTILE / 4); i += 32) {   // 256 float4s / 32 lanes
        int t4 = (i & 3) << 2;
        int cc = i >> 2;
        v4f q4 = *(const v4f*)(qg + (size_t)cc * T + t0w + t4);
#pragma unroll
        for (int j = 0; j < 4; ++j) Qs[w][t4 + j][cc] = (__bf16)(q4[j] * SCALE2);
    }
    __syncthreads();

    const int row16 = lane & 15;
    const int hi16  = lane >> 4;   // K-half selector
    const int off8  = hi16 * 8;    // C-layout: lanes>=16 hold rows 8..15
    const int off16 = hi16 * 16;   // B-layout: lanes>=16 hold K 16..31

    // Q as loop-invariant *B* operands (S^T = K^T x Q): B[k=c][n=t] from Qs[t][c]
    const v16bf bq0 = lds_frag16(&Qs[w][row16][     off16]);   // c 0..31
    const v16bf bq1 = lds_frag16(&Qs[w][row16][32 + off16]);   // c 32..63

    v8f o0 = {}, o1 = {}, o2 = {}, o3 = {};   // O^T accumulators (16 t x 64 c)
    float mlane = -1e30f;   // softmax state per t-column (t = lane&15, dup'd in halves)
    float llane = 0.f;

    for (int s0 = 0; s0 < S_TOT; s0 += SCHUNK) {
        __syncthreads();  // prior Ks/Vt readers done

        // ---- cooperative stage (S_ENC == T == 1024, so row stride is uniform) ----
        const float* kbase = (s0 < S_ENC) ? (ekg + s0) : (kg + (s0 - S_ENC));
        const float* vbase = (s0 < S_ENC) ? (evg + s0) : (vg + (s0 - S_ENC));
        for (int i = tid; i < C * (SCHUNK / 4); i += BLOCK) {
            int s4 = (i & (SCHUNK / 4 - 1)) << 2;
            int cc = i >> 4;
            v4f k4 = *(const v4f*)(kbase + (size_t)cc * 1024 + s4);
            v4f vv = *(const v4f*)(vbase + (size_t)cc * 1024 + s4);
#pragma unroll
            for (int j = 0; j < 4; ++j) Ks[s4 + j][cc] = (__bf16)k4[j];  // transpose
            v4bf vp;
#pragma unroll
            for (int j = 0; j < 4; ++j) vp[j] = (__bf16)vv[j];
            *(v4bf*)&Vt[cc][s4] = vp;                                    // natural
        }
        __syncthreads();

        // ---- scores transposed: S^T = K^T Q, four 16s x 16t tiles ----
        // A = K^T: lane = s row, K = c (contiguous from Ks[s][c])
        v8f st[4];
#pragma unroll
        for (int j = 0; j < 4; ++j) {
            const __bf16* krow = &Ks[j * 16 + row16][0];
            v8f acc = {};
            acc = wmma_bf16(lds_frag(krow,      off16, off16 + 8), bq0, acc);
            acc = wmma_bf16(lds_frag(krow + 32, off16, off16 + 8), bq1, acc);
            st[j] = acc;   // C-layout: row = s (r + 8*hi16), col = t (lane&15)
        }

        // ---- online softmax (exp2 domain): s-axis in-lane + one half-swap ----
        float tmax = -1e30f;
#pragma unroll
        for (int j = 0; j < 4; ++j)
#pragma unroll
            for (int r = 0; r < 8; ++r) tmax = fmaxf(tmax, st[j][r]);
        tmax = fmaxf(tmax, __shfl_xor(tmax, 16, 32));
        float mn    = fmaxf(mlane, tmax);
        float alpha = fast_exp2(mlane - mn);
        mlane = mn;

        float psum = 0.f;
#pragma unroll
        for (int j = 0; j < 4; ++j) {
            v8bf pp;
#pragma unroll
            for (int r = 0; r < 8; ++r) {
                float p = fast_exp2(st[j][r] - mn);
                psum += p;
                pp[r] = (__bf16)p;
            }
            // lane t holds s = j*16 + 8*hi16 + (0..7): contiguous -> one b128 store
            *(v8bf*)&Ps[w][row16][j * 16 + off8] = pp;
        }
        psum += __shfl_xor(psum, 16, 32);
        llane = llane * alpha + psum;

        // rescale O rows (row t = r + 8*hi16): broadcast alpha from lane t
#pragma unroll
        for (int r = 0; r < 8; ++r) {
            float ar = __shfl(alpha, r + (hi16 << 3), 32);
            o0[r] *= ar; o1[r] *= ar; o2[r] *= ar; o3[r] *= ar;
        }
        __syncthreads();  // P visible (uniform across waves)

        // ---- O^T += P (16x64) x V^T (64x64): 4 c-subtiles x 2 K-chunks ----
        const v16bf ap0 = lds_frag(&Ps[w][row16][0],  off16, off16 + 8);  // s 0..31
        const v16bf ap1 = lds_frag(&Ps[w][row16][32], off16, off16 + 8);  // s 32..63
        o0 = wmma_bf16(ap0, lds_frag16(&Vt[ 0 + row16][     off16]), o0);
        o0 = wmma_bf16(ap1, lds_frag16(&Vt[ 0 + row16][32 + off16]), o0);
        o1 = wmma_bf16(ap0, lds_frag16(&Vt[16 + row16][     off16]), o1);
        o1 = wmma_bf16(ap1, lds_frag16(&Vt[16 + row16][32 + off16]), o1);
        o2 = wmma_bf16(ap0, lds_frag16(&Vt[32 + row16][     off16]), o2);
        o2 = wmma_bf16(ap1, lds_frag16(&Vt[32 + row16][32 + off16]), o2);
        o3 = wmma_bf16(ap0, lds_frag16(&Vt[48 + row16][     off16]), o3);
        o3 = wmma_bf16(ap1, lds_frag16(&Vt[48 + row16][32 + off16]), o3);
    }

    // ---- epilogue: normalize rows (broadcast 1/l from lane t), scatter fp32 ----
    float linv = __builtin_amdgcn_rcpf(llane);
#pragma unroll
    for (int r = 0; r < 8; ++r) {
        float inv = __shfl(linv, r + (hi16 << 3), 32);
        int tt = t0w + r + (hi16 << 3);
        og[(size_t)( 0 + row16) * T + tt] = o0[r] * inv;
        og[(size_t)(16 + row16) * T + tt] = o1[r] * inv;
        og[(size_t)(32 + row16) * T + tt] = o2[r] * inv;
        og[(size_t)(48 + row16) * T + tt] = o3[r] * inv;
    }
}

extern "C" void kernel_launch(void* const* d_in, const int* in_sizes, int n_in,
                              void* d_out, int out_size, void* d_ws, size_t ws_size,
                              hipStream_t stream) {
    (void)in_sizes; (void)n_in; (void)out_size; (void)d_ws; (void)ws_size;
    const float* x   = (const float*)d_in[0];   // (4, 1536, 1024) fp32
    const float* ekv = (const float*)d_in[1];   // (4, 1024, 1024) fp32
    float*       out = (float*)d_out;           // (4,  512, 1024) fp32

    dim3 grid(T / TBLK, 4 * HEADS);  // 16 t-blocks x 32 batch-heads
    dim3 block(BLOCK);
    qkv_attn_flash_bf16<<<grid, block, 0, stream>>>(x, ekv, out);
}
